// ShmmLm_53102975648282
// MI455X (gfx1250) — matrix operand order repaired and checked by reading, not compiled, not run on hardware
//
#include <hip/hip_runtime.h>
#include <math.h>

// Problem dims (fixed by reference)
#define C_DIM 8192
#define H_DIM 256
#define V_DIM 10000
#define S_DIM 64
#define B_DIM 32
#define T_DIM 256

typedef __attribute__((ext_vector_type(16))) __bf16 v16bf;
typedef __attribute__((ext_vector_type(8)))  __bf16 v8bf;
typedef __attribute__((ext_vector_type(8)))  float  v8f;

// ---------- monotonic float<->uint key for atomic float max ----------
__device__ __forceinline__ unsigned fkey(float f) {
  unsigned u = __float_as_uint(f);
  return (u & 0x80000000u) ? ~u : (u | 0x80000000u);
}
__device__ __forceinline__ float fkey_inv(unsigned k) {
  unsigned u = (k & 0x80000000u) ? (k & 0x7FFFFFFFu) : ~k;
  return __uint_as_float(u);
}

// ---------- fp32 -> bf16 convert ----------
__global__ void f32_to_bf16_kernel(const float* __restrict__ in, __bf16* __restrict__ out, long n) {
  long i = (long)blockIdx.x * blockDim.x + threadIdx.x;
  long stride = (long)gridDim.x * blockDim.x;
  for (; i < n; i += stride) out[i] = (__bf16)in[i];
}

// ---------- WMMA GEMM: out = epilogue(A[MxK]bf16 @ W[KxN]bf16 + bias) ----------
// Block: 128x128 tile, 256 threads = 8 waves (2x4), wave tile 64x32 (4x2 wmma tiles).
template<bool RELU, bool RES, bool OUT_BF, bool OUT_F32>
__global__ __launch_bounds__(256, 2)
void wmma_gemm_kernel(const __bf16* __restrict__ A, const __bf16* __restrict__ W,
                      const float* __restrict__ bias, const float* __restrict__ res,
                      __bf16* __restrict__ outb, float* __restrict__ outf,
                      int M, int N, int K)
{
  __shared__ __align__(32) __bf16 Alds[128 * 32];        // row-major MxK tile
  __shared__ __align__(32) __bf16 Blds[128 * 32];        // transposed: [n][k]
  const int tid   = threadIdx.x;
  const int lane  = tid & 31;
  const int wave  = tid >> 5;
  const int waveM = wave >> 2;     // 0..1
  const int waveN = wave & 3;      // 0..3
  const int hl    = lane >> 4;     // half-wave select
  const int l15   = lane & 15;
  const int rowBase = blockIdx.y * 128;
  const int colBase = blockIdx.x * 128;

  v8f acc[4][2];
  #pragma unroll
  for (int mt = 0; mt < 4; ++mt)
    #pragma unroll
    for (int nt = 0; nt < 2; ++nt)
      #pragma unroll
      for (int r = 0; r < 8; ++r)
        acc[mt][nt][r] = 0.0f;

  for (int kk = 0; kk < K; kk += 32) {
    // Stage A (128x32 bf16): 512 chunks of 8 bf16, 2 per thread, 16B vector loads
    #pragma unroll
    for (int it = 0; it < 2; ++it) {
      int idx = tid + it * 256;        // 0..511
      int r = idx >> 2;                // row 0..127
      int cc = idx & 3;                // 8-elem chunk 0..3
      *(v8bf*)(&Alds[r * 32 + cc * 8]) =
          *(const v8bf*)(A + (size_t)(rowBase + r) * K + kk + cc * 8);
    }
    // Stage B transposed (Blds[n][k] = W[kk+k][colBase+n]); global reads coalesced over n
    #pragma unroll
    for (int it = 0; it < 16; ++it) {
      int idx = tid + it * 256;        // 0..4095
      int k = idx >> 7;                // 0..31
      int n = idx & 127;               // 0..127
      Blds[n * 32 + k] = W[(size_t)(kk + k) * N + colBase + n];
    }
    __syncthreads();

    // A fragment per ISA 16-bit 16x32 layout:
    //   lanes 0-15 : K = 0..7 (v0..3), 16..23 (v4..7);  lanes 16-31: +8
    v16bf afrag[4];
    #pragma unroll
    for (int mt = 0; mt < 4; ++mt) {
      const __bf16* pa = &Alds[(waveM * 64 + mt * 16 + l15) * 32 + hl * 8];
      v8bf lo = *(const v8bf*)pa;          // K = hl*8 .. +7
      v8bf hi = *(const v8bf*)(pa + 16);   // K = 16+hl*8 .. +7
      afrag[mt] = __builtin_shufflevector(lo, hi, 0,1,2,3,4,5,6,7,8,9,10,11,12,13,14,15);
    }
    // B fragment: lane holds column n, K = hl*8 .. +7, 16+hl*8 .. +7 (same as A)
    v16bf bfrag[2];
    #pragma unroll
    for (int nt = 0; nt < 2; ++nt) {
      const __bf16* pb = &Blds[(waveN * 32 + nt * 16 + l15) * 32 + hl * 8];
      v8bf blo = *(const v8bf*)pb;
      v8bf bhi = *(const v8bf*)(pb + 16);
      bfrag[nt] = __builtin_shufflevector(blo, bhi, 0,1,2,3,4,5,6,7,8,9,10,11,12,13,14,15);
    }
    #pragma unroll
    for (int mt = 0; mt < 4; ++mt)
      #pragma unroll
      for (int nt = 0; nt < 2; ++nt)
        acc[mt][nt] = __builtin_amdgcn_wmma_f32_16x16x32_bf16(
            false, afrag[mt], false, bfrag[nt], (short)0, acc[mt][nt], false, false);
    __syncthreads();
  }

  // Epilogue. C layout: VGPR r -> row = r + hl*8, col = l15.
  #pragma unroll
  for (int mt = 0; mt < 4; ++mt)
    #pragma unroll
    for (int nt = 0; nt < 2; ++nt)
      #pragma unroll
      for (int r = 0; r < 8; ++r) {
        int row = rowBase + waveM * 64 + mt * 16 + hl * 8 + r;
        int col = colBase + waveN * 32 + nt * 16 + l15;
        float v = acc[mt][nt][r] + bias[col];
        if (RELU) v = fmaxf(v, 0.0f);
        if (RES)  v += res[(size_t)row * N + col];
        if (OUT_BF)  outb[(size_t)row * N + col] = (__bf16)v;
        if (OUT_F32) outf[(size_t)row * N + col] = v;
      }
}

// ---------- start logits: out[c] = h2[c,:] . Wo[:,0] + bo ----------
__global__ __launch_bounds__(256)
void start_logits_kernel(const float* __restrict__ h2, const float* __restrict__ Wo,
                         const float* __restrict__ bo, float* __restrict__ out) {
  int c = blockIdx.x * 256 + threadIdx.x;
  if (c >= C_DIM) return;
  float s = bo[0];
  #pragma unroll 4
  for (int k = 0; k < H_DIM; ++k) s += h2[(size_t)c * H_DIM + k] * Wo[k];
  out[c] = s;
}

// ---------- log_softmax over a length-n vector (single block) ----------
__global__ __launch_bounds__(1024)
void vec_log_softmax_kernel(float* __restrict__ x, int n) {
  __shared__ float red[1024];
  int tid = threadIdx.x;
  float m = -INFINITY;
  for (int i = tid; i < n; i += 1024) m = fmaxf(m, x[i]);
  red[tid] = m; __syncthreads();
  for (int st = 512; st; st >>= 1) { if (tid < st) red[tid] = fmaxf(red[tid], red[tid + st]); __syncthreads(); }
  float M = red[0]; __syncthreads();
  float s = 0.f;
  for (int i = tid; i < n; i += 1024) s += expf(x[i] - M);
  red[tid] = s; __syncthreads();
  for (int st = 512; st; st >>= 1) { if (tid < st) red[tid] += red[tid + st]; __syncthreads(); }
  float lse = M + logf(red[0]);
  __syncthreads();
  for (int i = tid; i < n; i += 1024) x[i] -= lse;
}

// ---------- per-row log_softmax of transition logits; last col forced -inf ----------
__global__ __launch_bounds__(256)
void row_log_softmax_kernel(float* __restrict__ tl) {
  __shared__ float red[256];
  int row = blockIdx.x, tid = threadIdx.x;
  float* p = tl + (size_t)row * C_DIM;
  float m = -INFINITY;
  for (int c = tid; c < C_DIM - 1; c += 256) m = fmaxf(m, p[c]);
  red[tid] = m; __syncthreads();
  for (int st = 128; st; st >>= 1) { if (tid < st) red[tid] = fmaxf(red[tid], red[tid + st]); __syncthreads(); }
  float M = red[0]; __syncthreads();
  float s = 0.f;
  for (int c = tid; c < C_DIM - 1; c += 256) s += expf(p[c] - M);
  red[tid] = s; __syncthreads();
  for (int st = 128; st; st >>= 1) { if (tid < st) red[tid] += red[tid + st]; __syncthreads(); }
  float lse = M + logf(red[0]);
  __syncthreads();
  for (int c = tid; c < C_DIM; c += 256) p[c] = (c == C_DIM - 1) ? -INFINITY : p[c] - lse;
}

// ---------- Wo transpose for the emission branch: WoT[v][k] = Wo[k][v] ----------
__global__ __launch_bounds__(256)
void transpose_wo_kernel(const float* __restrict__ Wo, float* __restrict__ WoT) {
  __shared__ float tile[32][33];
  int vb = blockIdx.x * 32, kb = blockIdx.y * 32;
  int tx = threadIdx.x & 31, ty = threadIdx.x >> 5;   // ty = 0..7
  #pragma unroll
  for (int yy = 0; yy < 32; yy += 8) {
    int k = kb + ty + yy, v = vb + tx;
    tile[ty + yy][tx] = (v < V_DIM) ? Wo[(size_t)k * V_DIM + v] : 0.f;
  }
  __syncthreads();
  #pragma unroll
  for (int yy = 0; yy < 32; yy += 8) {
    int v = vb + ty + yy, k = kb + tx;
    if (v < V_DIM) WoT[(size_t)v * H_DIM + k] = tile[tx][ty + yy];
  }
}

// ---------- emission: sparse logits only at masked (state,word) pairs ----------
__global__ void init_em_kernel(unsigned* __restrict__ mbits, float* __restrict__ dsum) {
  int c = blockIdx.x * 256 + threadIdx.x;
  if (c < C_DIM) { mbits[c] = 0u; dsum[c] = 0.f; }
}

__global__ __launch_bounds__(256)
void em_pass1_kernel(const int* __restrict__ w2s, const float* __restrict__ h2,
                     const float* __restrict__ WoT, const float* __restrict__ bo,
                     float* __restrict__ Eval, unsigned* __restrict__ mbits) {
  int idx = blockIdx.x * 256 + threadIdx.x;     // (v,s) pair
  if (idx >= V_DIM * S_DIM) return;
  int v = idx / S_DIM;
  int c = w2s[idx];
  const float* hr = h2 + (size_t)c * H_DIM;
  const float* wr = WoT + (size_t)v * H_DIM;    // contiguous, broadcast across lanes
  float acc = bo[v];
  #pragma unroll 8
  for (int k = 0; k < H_DIM; ++k) acc += hr[k] * wr[k];
  Eval[idx] = acc;
  atomicMax(&mbits[c], fkey(acc));              // max is idempotent: duplicates harmless
}

__global__ __launch_bounds__(256)
void em_pass2_kernel(const int* __restrict__ w2s, const float* __restrict__ Eval,
                     const unsigned* __restrict__ mbits, float* __restrict__ dsum) {
  int idx = blockIdx.x * 256 + threadIdx.x;
  if (idx >= V_DIM * S_DIM) return;
  int v = idx / S_DIM, s = idx % S_DIM;
  int c = w2s[idx];
  for (int s2 = 0; s2 < s; ++s2)
    if (w2s[v * S_DIM + s2] == c) return;       // mask has set semantics: dedupe
  atomicAdd(&dsum[c], expf(Eval[idx] - fkey_inv(mbits[c])));
}

__global__ void em_pass3_kernel(const unsigned* __restrict__ mbits, const float* __restrict__ dsum,
                                float* __restrict__ dlse) {
  int c = blockIdx.x * 256 + threadIdx.x;
  if (c >= C_DIM) return;
  float s = dsum[c];
  dlse[c] = (s > 0.f) ? fkey_inv(mbits[c]) + logf(s) : -INFINITY;
}

__global__ void obs_kernel(const int* __restrict__ text, const int* __restrict__ w2s,
                           const float* __restrict__ Eval, const float* __restrict__ dlse,
                           float* __restrict__ obs) {
  int idx = blockIdx.x * 256 + threadIdx.x;     // b*T*S layout
  if (idx >= B_DIM * T_DIM * S_DIM) return;
  int s = idx % S_DIM, bt = idx / S_DIM;
  int v = text[bt];
  int c = w2s[v * S_DIM + s];
  obs[idx] = Eval[(size_t)v * S_DIM + s] - dlse[c];
}

// ---------- P[t,b,i,j] = trans[cs[b,t,j], cs[b,t+1,i]] + obs[b,t+1,i] (+ init at t=0)
//            Also writes PT[t,b,j,i] (tile-transposed) for a coalesced backward scan.
__global__ __launch_bounds__(256)
void build_p_kernel(const int* __restrict__ text, const int* __restrict__ w2s,
                    const float* __restrict__ trans, const float* __restrict__ obs,
                    const float* __restrict__ startn, float* __restrict__ P,
                    float* __restrict__ PT) {
  long idx = (long)blockIdx.x * 256 + threadIdx.x;
  const long NP = (long)(T_DIM - 1) * B_DIM * S_DIM * S_DIM;
  if (idx >= NP) return;
  int j = (int)(idx & (S_DIM - 1));
  int i = (int)((idx >> 6) & (S_DIM - 1));
  int b = (int)((idx >> 12) & (B_DIM - 1));
  int t = (int)(idx >> 17);
  int vt = text[b * T_DIM + t];
  int vn = text[b * T_DIM + t + 1];
  int cp = w2s[vt * S_DIM + j];
  int cn = w2s[vn * S_DIM + i];
  float val = trans[(size_t)cp * C_DIM + cn] + obs[(b * T_DIM + t + 1) * S_DIM + i];
  if (t == 0) val += startn[cp] + obs[(b * T_DIM) * S_DIM + j];
  P[idx] = val;
  PT[(((size_t)t * B_DIM + b) * S_DIM + j) * S_DIM + i] = val;
}

// ---------- forward scan (256 thr/block): a_new[i] = lse_j(P[t,b,i,j] + a[j]) ----------
// thread = (i = tid&63, q = tid>>6); each does a 16-wide partial online LSE, then 4-way combine.
__global__ __launch_bounds__(256)
void forward_scan_kernel(const float* __restrict__ P, float* __restrict__ alphas,
                         float* __restrict__ Zb, int Tm1) {
  int b = blockIdx.x, tid = threadIdx.x;
  int i = tid & 63, q = tid >> 6;
  __shared__ float a[S_DIM];
  __shared__ float pm[4][S_DIM + 1];
  __shared__ float ps[4][S_DIM + 1];
  if (tid < S_DIM) a[tid] = 0.f;
  __syncthreads();
  for (int t = 0; t < Tm1; ++t) {
    const float* Pt = P + (((size_t)t * B_DIM + b) * S_DIM + i) * S_DIM + q * 16;
    if (t + 1 < Tm1) {
      // hide next step's L2/HBM latency under this step's exp chain
      const float* Pn = P + ((size_t)(t + 1) * B_DIM + b) * S_DIM * S_DIM;
      __builtin_prefetch(Pn + tid * 16, 0, 3);
    }
    float m = -INFINITY, s = 0.f;
    #pragma unroll
    for (int jj = 0; jj < 16; ++jj) {
      float x = Pt[jj] + a[q * 16 + jj];
      if (x == -INFINITY) continue;
      if (x > m) { s = s * __expf(m - x) + 1.f; m = x; }
      else       { s += __expf(x - m); }
    }
    pm[q][i] = m; ps[q][i] = s;
    __syncthreads();
    if (tid < S_DIM) {
      float M = fmaxf(fmaxf(pm[0][tid], pm[1][tid]), fmaxf(pm[2][tid], pm[3][tid]));
      float S = 0.f;
      #pragma unroll
      for (int qq = 0; qq < 4; ++qq)
        if (pm[qq][tid] != -INFINITY) S += ps[qq][tid] * __expf(pm[qq][tid] - M);
      float anew = (S > 0.f) ? M + __logf(S) : -INFINITY;
      a[tid] = anew;
      alphas[((size_t)t * B_DIM + b) * S_DIM + tid] = anew;
    }
    __syncthreads();
  }
  if (tid == 0) {
    float m = -INFINITY;
    for (int jj = 0; jj < S_DIM; ++jj) m = fmaxf(m, a[jj]);
    float s = 0.f;
    for (int jj = 0; jj < S_DIM; ++jj) s += __expf(a[jj] - m);
    Zb[b] = m + __logf(s);
  }
}

// ---------- backward scan on PT: b_prev[j] = lse_i(PT[t,b,j,i] + bn[i]); emit betas[t]=bn ----------
__global__ __launch_bounds__(256)
void backward_scan_kernel(const float* __restrict__ PT, float* __restrict__ betas, int Tm1) {
  int b = blockIdx.x, tid = threadIdx.x;
  int j = tid & 63, q = tid >> 6;
  __shared__ float bn[S_DIM];
  __shared__ float pm[4][S_DIM + 1];
  __shared__ float ps[4][S_DIM + 1];
  if (tid < S_DIM) bn[tid] = 0.f;
  __syncthreads();
  for (int t = Tm1 - 1; t >= 0; --t) {
    if (tid < S_DIM) betas[((size_t)t * B_DIM + b) * S_DIM + tid] = bn[tid];  // emit carry-in
    const float* Pt = PT + (((size_t)t * B_DIM + b) * S_DIM + j) * S_DIM + q * 16;
    if (t > 0) {
      const float* Pn = PT + ((size_t)(t - 1) * B_DIM + b) * S_DIM * S_DIM;
      __builtin_prefetch(Pn + tid * 16, 0, 3);
    }
    float m = -INFINITY, s = 0.f;
    #pragma unroll
    for (int ii = 0; ii < 16; ++ii) {
      float x = Pt[ii] + bn[q * 16 + ii];
      if (x == -INFINITY) continue;
      if (x > m) { s = s * __expf(m - x) + 1.f; m = x; }
      else       { s += __expf(x - m); }
    }
    pm[q][j] = m; ps[q][j] = s;
    __syncthreads();
    if (tid < S_DIM) {
      float M = fmaxf(fmaxf(pm[0][tid], pm[1][tid]), fmaxf(pm[2][tid], pm[3][tid]));
      float S = 0.f;
      #pragma unroll
      for (int qq = 0; qq < 4; ++qq)
        if (pm[qq][tid] != -INFINITY) S += ps[qq][tid] * __expf(pm[qq][tid] - M);
      bn[tid] = (S > 0.f) ? M + __logf(S) : -INFINITY;
    }
    __syncthreads();
  }
}

// ---------- elbo partial sums (deterministic block tree) ----------
__global__ __launch_bounds__(256)
void elbo_partial_kernel(const float* __restrict__ P, const float* __restrict__ alphas,
                         const float* __restrict__ betas, const float* __restrict__ Zb,
                         float* __restrict__ partials) {
  __shared__ float red[256];
  long idx = (long)blockIdx.x * 256 + threadIdx.x;
  const long NP = (long)(T_DIM - 1) * B_DIM * S_DIM * S_DIM;
  float contrib = 0.f;
  if (idx < NP) {
    int j = (int)(idx & 63), i = (int)((idx >> 6) & 63);
    int b = (int)((idx >> 12) & 31), t = (int)(idx >> 17);
    float p = P[idx];
    if (isfinite(p)) {
      float ap = (t == 0) ? 0.f : alphas[((size_t)(t - 1) * B_DIM + b) * S_DIM + j];
      float be = betas[((size_t)t * B_DIM + b) * S_DIM + i];
      contrib = expf(ap + p + be - Zb[b]) * p;
    }
  }
  red[threadIdx.x] = contrib; __syncthreads();
  for (int st = 128; st; st >>= 1) { if (threadIdx.x < st) red[threadIdx.x] += red[threadIdx.x + st]; __syncthreads(); }
  if (threadIdx.x == 0) partials[blockIdx.x] = red[0];
}

// ---------- final deterministic reduction: out = [evidence, elbo] ----------
__global__ __launch_bounds__(256)
void finalize_kernel(const float* __restrict__ Zb, const float* __restrict__ partials,
                     int npart, float* __restrict__ out) {
  __shared__ float red[256];
  int tid = threadIdx.x;
  float s = 0.f;
  for (int i = tid; i < npart; i += 256) s += partials[i];
  red[tid] = s; __syncthreads();
  for (int st = 128; st; st >>= 1) { if (tid < st) red[tid] += red[tid + st]; __syncthreads(); }
  if (tid == 0) {
    float ev = 0.f;
    for (int b = 0; b < B_DIM; ++b) ev += Zb[b];
    out[0] = ev;
    out[1] = red[0];
  }
}

extern "C" void kernel_launch(void* const* d_in, const int* in_sizes, int n_in,
                              void* d_out, int out_size, void* d_ws, size_t ws_size,
                              hipStream_t stream) {
  (void)in_sizes; (void)n_in; (void)out_size; (void)ws_size;
  // JAX tree-flatten order (dict keys sorted): params{pre_emb, start_emb,
  // start_mlp{W1,W2,Wo,b1,b2,bo}, state_emb, term_mlp{...}, trans_mlp{...}}, text, word2state
  const float* pre_emb   = (const float*)d_in[0];
  const float* start_emb = (const float*)d_in[1];
  const float* sW1 = (const float*)d_in[2];
  const float* sW2 = (const float*)d_in[3];
  const float* sWo = (const float*)d_in[4];
  const float* sb1 = (const float*)d_in[5];
  const float* sb2 = (const float*)d_in[6];
  const float* sbo = (const float*)d_in[7];
  const float* state_emb = (const float*)d_in[8];
  const float* eW1 = (const float*)d_in[9];    // term_mlp
  const float* eW2 = (const float*)d_in[10];
  const float* eWo = (const float*)d_in[11];
  const float* eb1 = (const float*)d_in[12];
  const float* eb2 = (const float*)d_in[13];
  const float* ebo = (const float*)d_in[14];
  const float* rW1 = (const float*)d_in[15];   // trans_mlp
  const float* rW2 = (const float*)d_in[16];
  const float* rWo = (const float*)d_in[17];
  const float* rb1 = (const float*)d_in[18];
  const float* rb2 = (const float*)d_in[19];
  const float* rbo = (const float*)d_in[20];
  const int* text = (const int*)d_in[21];
  const int* w2s  = (const int*)d_in[22];
  float* out = (float*)d_out;

  char* ws = (char*)d_ws;
  size_t off = 0;
  auto take = [&](size_t bytes) -> void* {
    void* p = ws + off;
    off += (bytes + 255) & ~(size_t)255;
    return p;
  };

  float*    trans  = (float*)   take((size_t)C_DIM * C_DIM * 4);                       // 256 MB
  float*    P      = (float*)   take((size_t)(T_DIM - 1) * B_DIM * S_DIM * S_DIM * 4); // 134 MB
  float*    PT     = (float*)   take((size_t)(T_DIM - 1) * B_DIM * S_DIM * S_DIM * 4); // 134 MB
  __bf16*   xbf    = (__bf16*)  take((size_t)C_DIM * H_DIM * 2);
  __bf16*   h1bf   = (__bf16*)  take((size_t)C_DIM * H_DIM * 2);
  __bf16*   h2bf   = (__bf16*)  take((size_t)C_DIM * H_DIM * 2);
  __bf16*   w1bf   = (__bf16*)  take((size_t)H_DIM * H_DIM * 2);
  __bf16*   w2bf   = (__bf16*)  take((size_t)H_DIM * H_DIM * 2);
  __bf16*   wobf   = (__bf16*)  take((size_t)H_DIM * C_DIM * 2);
  float*    h2sf   = (float*)   take((size_t)C_DIM * H_DIM * 4);
  float*    h2pf   = (float*)   take((size_t)C_DIM * H_DIM * 4);
  float*    wot    = (float*)   take((size_t)V_DIM * H_DIM * 4);                        // 10 MB
  float*    slog   = (float*)   take((size_t)C_DIM * 4);
  float*    Eval   = (float*)   take((size_t)V_DIM * S_DIM * 4);
  unsigned* mbits  = (unsigned*)take((size_t)C_DIM * 4);
  float*    dsum   = (float*)   take((size_t)C_DIM * 4);
  float*    dlse   = (float*)   take((size_t)C_DIM * 4);
  float*    obs    = (float*)   take((size_t)B_DIM * T_DIM * S_DIM * 4);
  float*    alphas = (float*)   take((size_t)(T_DIM - 1) * B_DIM * S_DIM * 4);
  float*    betas  = (float*)   take((size_t)(T_DIM - 1) * B_DIM * S_DIM * 4);
  float*    Zb     = (float*)   take((size_t)B_DIM * 4);
  const long NP = (long)(T_DIM - 1) * B_DIM * S_DIM * S_DIM;
  const int nPart = (int)(NP / 256);                                                   // exact
  float*    partials = (float*) take((size_t)nPart * 4);

  dim3 gH(H_DIM / 128, C_DIM / 128);   // 8192x256 output
  dim3 gT(C_DIM / 128, C_DIM / 128);   // 8192x8192 output

  auto conv = [&](const float* src, __bf16* dst, long n) {
    int blocks = (int)((n + 255) / 256);
    if (blocks > 4096) blocks = 4096;
    f32_to_bf16_kernel<<<blocks, 256, 0, stream>>>(src, dst, n);
  };

  // ---- start branch: log_softmax(res_mlp(start_emb)[:,0]) ----
  conv(start_emb, xbf, (long)C_DIM * H_DIM);
  conv(sW1, w1bf, H_DIM * H_DIM);
  conv(sW2, w2bf, H_DIM * H_DIM);
  wmma_gemm_kernel<true, false, true, false><<<gH, 256, 0, stream>>>(
      xbf, w1bf, sb1, nullptr, h1bf, nullptr, C_DIM, H_DIM, H_DIM);
  wmma_gemm_kernel<true, true, false, true><<<gH, 256, 0, stream>>>(
      h1bf, w2bf, sb2, start_emb, nullptr, h2sf, C_DIM, H_DIM, H_DIM);
  start_logits_kernel<<<C_DIM / 256, 256, 0, stream>>>(h2sf, sWo, sbo, slog);
  vec_log_softmax_kernel<<<1, 1024, 0, stream>>>(slog, C_DIM);

  // ---- transition branch: log_softmax(res_mlp(state_emb)@Wo, last col -inf) ----
  conv(state_emb, xbf, (long)C_DIM * H_DIM);
  conv(rW1, w1bf, H_DIM * H_DIM);
  conv(rW2, w2bf, H_DIM * H_DIM);
  conv(rWo, wobf, (long)H_DIM * C_DIM);
  wmma_gemm_kernel<true, false, true, false><<<gH, 256, 0, stream>>>(
      xbf, w1bf, rb1, nullptr, h1bf, nullptr, C_DIM, H_DIM, H_DIM);
  wmma_gemm_kernel<true, true, true, false><<<gH, 256, 0, stream>>>(
      h1bf, w2bf, rb2, state_emb, h2bf, nullptr, C_DIM, H_DIM, H_DIM);
  wmma_gemm_kernel<false, false, false, true><<<gT, 256, 0, stream>>>(
      h2bf, wobf, rbo, nullptr, nullptr, trans, C_DIM, C_DIM, H_DIM);
  row_log_softmax_kernel<<<C_DIM, 256, 0, stream>>>(trans);

  // ---- emission branch (sparse: only masked (state,word) pairs) ----
  conv(pre_emb, xbf, (long)C_DIM * H_DIM);
  conv(eW1, w1bf, H_DIM * H_DIM);
  conv(eW2, w2bf, H_DIM * H_DIM);
  wmma_gemm_kernel<true, false, true, false><<<gH, 256, 0, stream>>>(
      xbf, w1bf, eb1, nullptr, h1bf, nullptr, C_DIM, H_DIM, H_DIM);
  wmma_gemm_kernel<true, true, false, true><<<gH, 256, 0, stream>>>(
      h1bf, w2bf, eb2, pre_emb, nullptr, h2pf, C_DIM, H_DIM, H_DIM);
  transpose_wo_kernel<<<dim3((V_DIM + 31) / 32, H_DIM / 32), 256, 0, stream>>>(eWo, wot);
  init_em_kernel<<<C_DIM / 256, 256, 0, stream>>>(mbits, dsum);
  int vsBlocks = (V_DIM * S_DIM + 255) / 256;
  em_pass1_kernel<<<vsBlocks, 256, 0, stream>>>(w2s, h2pf, wot, ebo, Eval, mbits);
  em_pass2_kernel<<<vsBlocks, 256, 0, stream>>>(w2s, Eval, mbits, dsum);
  em_pass3_kernel<<<C_DIM / 256, 256, 0, stream>>>(mbits, dsum, dlse);
  obs_kernel<<<(B_DIM * T_DIM * S_DIM) / 256, 256, 0, stream>>>(text, w2s, Eval, dlse, obs);

  // ---- P tensor (+transposed copy), forward/backward scans, evidence + elbo ----
  build_p_kernel<<<nPart, 256, 0, stream>>>(text, w2s, trans, obs, slog, P, PT);
  forward_scan_kernel<<<B_DIM, 256, 0, stream>>>(P, alphas, Zb, T_DIM - 1);
  backward_scan_kernel<<<B_DIM, 256, 0, stream>>>(PT, betas, T_DIM - 1);
  elbo_partial_kernel<<<nPart, 256, 0, stream>>>(P, alphas, betas, Zb, partials);
  finalize_kernel<<<1, 256, 0, stream>>>(Zb, partials, nPart, out);
}
